// PeepholeLSTMModule_11759620456859
// MI455X (gfx1250) — matrix operand order repaired and checked by reading, not compiled, not run on hardware
//
#include <hip/hip_runtime.h>
#include <hip/hip_bf16.h>

// ---------------------------------------------------------------------------
// Bidirectional peephole LSTM for MI455X (gfx1250), wave32 + WMMA + async LDS.
//   B=8, T=2048, D=256, H=256, 4H=1024.
// Phase 1: xW = x @ W + b (both dirs) -> fp32 workspace (128 MB), WMMA f16.
// Phase 2: sequential scan, 1 workgroup/direction. U register-resident as f16
//          WMMA B-fragments; xW step tiles double-buffered in LDS via
//          global_load_async_to_lds_b128 (overlapped with compute).
// ---------------------------------------------------------------------------

typedef _Float16 half_t;
typedef __attribute__((ext_vector_type(16))) _Float16     v16h;
typedef __attribute__((ext_vector_type(8)))  float        v8f;
typedef __attribute__((ext_vector_type(8)))  unsigned int u32x8;

#define LSTM_B   8
#define LSTM_T   2048
#define LSTM_D   256
#define LSTM_H   256
#define LSTM_4H  1024
#define LSTM_M   (LSTM_B * LSTM_T)   // 16384 flattened rows

static __device__ __forceinline__ unsigned int pack2h(float a, float b) {
  union { _Float16 h[2]; unsigned int u; } p;
  p.h[0] = (_Float16)a;   // low 16 bits  = even K
  p.h[1] = (_Float16)b;   // high 16 bits = odd K
  return p.u;
}

static __device__ __forceinline__ float sigmoidf_(float x) {
  return 1.0f / (1.0f + __expf(-x));
}

// generic->LDS byte offset: low 32 bits of the generic address (aperture is in
// the high half per the CDNA5 flat-address aperture rules).
static __device__ __forceinline__ unsigned lds_offset(const void* p) {
  return (unsigned)(unsigned long long)p;
}

// async copy 16B global -> LDS (tracked by ASYNCcnt)
static __device__ __forceinline__ void async_g2l_b128(unsigned lds_off,
                                                      const float* gsrc) {
  asm volatile("global_load_async_to_lds_b128 %0, %1, off"
               :: "v"(lds_off), "v"(gsrc)
               : "memory");
}
static __device__ __forceinline__ void wait_async0() {
  asm volatile("s_wait_asynccnt 0x0" ::: "memory");
}

// ---------------------------------------------------------------------------
// Phase 1: xW = x @ W + bias, one direction per blockIdx.z.
// Workgroup: 256 threads (8 waves). Tile: 128 M x 64 N, K chunks of 32.
// LDS tiles padded to stride 20 dwords: 16B-aligned rows -> ds_load_b128
// fragment builds, conflict-free banks (20*L mod 64 distinct for L=0..15).
// ---------------------------------------------------------------------------
__global__ __launch_bounds__(256) void xw_gemm(
    const float* __restrict__ x,
    const float* __restrict__ Wf, const float* __restrict__ bf,
    const float* __restrict__ Wb, const float* __restrict__ bb,
    float* __restrict__ xw_ws)
{
  const int dir = blockIdx.z;
  const float* __restrict__ W    = dir ? Wb : Wf;
  const float* __restrict__ bias = dir ? bb : bf;
  float* __restrict__ xw = xw_ws + (size_t)dir * ((size_t)LSTM_M * LSTM_4H);

  const int mbase = blockIdx.y * 128;
  const int nbase = blockIdx.x * 64;

  __shared__ unsigned int Atile[128 * 20];   // [row][kpair], stride 20
  __shared__ unsigned int Btile[64 * 20];    // [col][kpair], stride 20

  const int tid  = threadIdx.x;
  const int wid  = tid >> 5;      // 0..7
  const int lane = tid & 31;
  const int l16  = lane & 15;
  const int half = lane >> 4;

  v8f acc[4];
  #pragma unroll
  for (int nt = 0; nt < 4; ++nt)
    #pragma unroll
    for (int r = 0; r < 8; ++r)
      acc[nt][r] = 0.0f;

  for (int kc = 0; kc < LSTM_D; kc += 32) {
    // ---- stage A: x[mbase..+128, kc..+32] -> f16 pairs
    #pragma unroll
    for (int q = 0; q < 8; ++q) {
      int lin = tid + q * 256;          // 0..2047
      int row = lin >> 4;
      int kp  = lin & 15;
      const float2 v =
          *reinterpret_cast<const float2*>(x + (size_t)(mbase + row) * LSTM_D + kc + 2 * kp);
      Atile[row * 20 + kp] = pack2h(v.x, v.y);
    }
    // ---- stage B: W[kc..+32, nbase..+64] -> f16 pairs, [col][kpair]
    #pragma unroll
    for (int q = 0; q < 4; ++q) {
      int lin = tid + q * 256;          // 0..1023
      int n   = lin & 63;
      int kp  = lin >> 6;
      int k   = kc + 2 * kp;
      float w0 = W[(size_t)k       * LSTM_4H + nbase + n];
      float w1 = W[(size_t)(k + 1) * LSTM_4H + nbase + n];
      Btile[n * 20 + kp] = pack2h(w0, w1);
    }
    __syncthreads();

    // ---- A fragment (16x32 f16): dword j<4 -> kpair j+4*half,
    //      j>=4 -> kpair 8+(j-4)+4*half  => two contiguous quads.
    const int arow = wid * 16 + l16;
    const uint4 a0 = *reinterpret_cast<const uint4*>(&Atile[arow * 20 + 4 * half]);
    const uint4 a1 = *reinterpret_cast<const uint4*>(&Atile[arow * 20 + 8 + 4 * half]);
    u32x8 au;
    au[0] = a0.x; au[1] = a0.y; au[2] = a0.z; au[3] = a0.w;
    au[4] = a1.x; au[5] = a1.y; au[6] = a1.z; au[7] = a1.w;
    const v16h afr = __builtin_bit_cast(v16h, au);

    // ---- 4 B fragments (32x16 f16): dword j<4 -> kpair j+4*half, j>=4 -> kpair 8+(j-4)+4*half
    #pragma unroll
    for (int nt = 0; nt < 4; ++nt) {
      const int bbase = (nt * 16 + l16) * 20 + 4 * half;
      const uint4 b0 = *reinterpret_cast<const uint4*>(&Btile[bbase]);
      const uint4 b1 = *reinterpret_cast<const uint4*>(&Btile[bbase + 8]);
      u32x8 bu;
      bu[0] = b0.x; bu[1] = b0.y; bu[2] = b0.z; bu[3] = b0.w;
      bu[4] = b1.x; bu[5] = b1.y; bu[6] = b1.z; bu[7] = b1.w;
      const v16h bfr = __builtin_bit_cast(v16h, bu);
      acc[nt] = __builtin_amdgcn_wmma_f32_16x16x32_f16(
          false, afr, false, bfr, (short)0, acc[nt], false, false);
    }
    __syncthreads();
  }

  // ---- epilogue: + bias, store fp32. C layout: VGPR r -> M = r + 8*half, N = l16.
  #pragma unroll
  for (int nt = 0; nt < 4; ++nt) {
    int col = nbase + nt * 16 + l16;
    float bv = bias[col];
    #pragma unroll
    for (int r = 0; r < 8; ++r) {
      int row = mbase + wid * 16 + r + 8 * half;
      xw[(size_t)row * LSTM_4H + col] = acc[nt][r] + bv;
    }
  }
}

// ---------------------------------------------------------------------------
// Phase 2: recurrent scan. grid(2) = direction, 1024 threads = 32 waves.
// Wave w owns columns [w*32, w*32+32). U (f16) register-resident as WMMA
// B-fragments (8 Kchunks x 2 Ntiles = 128 VGPRs/lane).
// Dynamic LDS: xwbuf[2][8][1024] fp32 (64 KB) double-buffers the per-step xW
// tile (async-copied) and doubles as the z scratch (wave-private columns).
// ---------------------------------------------------------------------------
__global__ __launch_bounds__(1024) void lstm_scan(
    const float* __restrict__ xw_ws,
    const float* __restrict__ Uf, const float* __restrict__ Ub,
    const float* __restrict__ pif, const float* __restrict__ pff,
    const float* __restrict__ pof,
    const float* __restrict__ pib, const float* __restrict__ pfb,
    const float* __restrict__ pob,
    float* __restrict__ out)
{
  const int dir = blockIdx.x;
  const float* __restrict__ U  = dir ? Ub  : Uf;
  const float* __restrict__ pi = dir ? pib : pif;
  const float* __restrict__ pf = dir ? pfb : pff;
  const float* __restrict__ po = dir ? pob : pof;
  const float* __restrict__ xw = xw_ws + (size_t)dir * ((size_t)LSTM_M * LSTM_4H);

  extern __shared__ float xwbuf[];                  // [2][8*1024] fp32, 64 KB
  // h as f16 pairs: 16 rows (8 batch + 8 zero-pad) x 128 K-pairs, stride 132
  // (528 B rows: 16B-aligned, banks 4*L -> conflict-free b128 loads).
  __shared__ unsigned int hbuf[16 * 132];
  __shared__ float        cbuf[LSTM_B * LSTM_H];    // 8 KB
  _Float16* hh = (_Float16*)hbuf;                   // half view: row stride 264

  const int tid  = threadIdx.x;
  const int wid  = tid >> 5;       // 0..31
  const int lane = tid & 31;
  const int l16  = lane & 15;
  const int half = lane >> 4;
  const int n0   = wid * 32;

  const unsigned xw_lds0 = lds_offset(&xwbuf[0]);
  const unsigned xw_ldssz = (unsigned)(LSTM_B * LSTM_4H * sizeof(float));

  // per-thread async-copy assignment: row r (8 rows), 32B segment within row
  const int arow = tid >> 7;                 // 0..7
  const int aseg = (tid & 127) * 8;          // float offset within row

  // zero-init h (incl. pad rows) and c
  for (int i = tid; i < 16 * 132; i += 1024) hbuf[i] = 0u;
  for (int i = tid; i < LSTM_B * LSTM_H; i += 1024) cbuf[i] = 0.0f;

  // ---- kick off async copy of xW[t0] into buffer 0
  {
    const int t0 = dir ? (LSTM_T - 1) : 0;
    const float* g = xw + ((size_t)(arow * LSTM_T + t0)) * LSTM_4H + aseg;
    const unsigned l = xw_lds0 + (unsigned)((arow * LSTM_4H + aseg) * 4);
    async_g2l_b128(l, g);
    async_g2l_b128(l + 16u, g + 4);
  }

  // ---- preload U as B-fragments (register resident: 16 x v16h = 128 VGPRs)
  v16h ufr[8][2];
  #pragma unroll
  for (int cc = 0; cc < 8; ++cc) {
    #pragma unroll
    for (int nt = 0; nt < 2; ++nt) {
      u32x8 bu;
      int n = n0 + nt * 16 + l16;
      #pragma unroll
      for (int j = 0; j < 8; ++j) {
        int k = cc * 32 + half * 8 + 2 * j + (j >= 4 ? 8 : 0);
        bu[j] = pack2h(U[(size_t)k       * LSTM_4H + n],
                       U[(size_t)(k + 1) * LSTM_4H + n]);
      }
      ufr[cc][nt] = __builtin_bit_cast(v16h, bu);
    }
  }

  // per-thread gate elements: e0 = tid, e1 = tid + 1024 -> (b = e>>8, j = e&255)
  const int j0 = tid & 255, b0 = tid >> 8;
  const int j1 = (tid + 1024) & 255, b1 = (tid + 1024) >> 8;
  const float pi0 = pi[j0], pf0 = pf[j0], po0 = po[j0];
  const float pi1 = pi[j1], pf1 = pf[j1], po1 = po[j1];

  wait_async0();
  __syncthreads();

  for (int s = 0; s < LSTM_T; ++s) {
    const int t   = dir ? (LSTM_T - 1 - s) : s;
    const int cur = s & 1;
    float* __restrict__ xwcur = xwbuf + (size_t)cur * (LSTM_B * LSTM_4H);

    // ---- issue async copy of next step's xW tile into the other buffer
    //      (overlaps the whole step; drained before the step-ending barrier)
    if (s + 1 < LSTM_T) {
      const int t2 = dir ? (LSTM_T - 2 - s) : (s + 1);
      const float* g = xw + ((size_t)(arow * LSTM_T + t2)) * LSTM_4H + aseg;
      const unsigned l = xw_lds0 + (unsigned)(cur ^ 1) * xw_ldssz
                       + (unsigned)((arow * LSTM_4H + aseg) * 4);
      async_g2l_b128(l, g);
      async_g2l_b128(l + 16u, g + 4);
    }

    // ---- C fragments = xW tile from LDS (pad rows wrap to valid batches)
    v8f acc[2];
    #pragma unroll
    for (int nt = 0; nt < 2; ++nt) {
      int col = n0 + nt * 16 + l16;
      #pragma unroll
      for (int r = 0; r < 8; ++r) {
        int b = (r + 8 * half) & 7;
        acc[nt][r] = xwcur[b * LSTM_4H + col];
      }
    }

    // ---- z = h @ U + xW : 8 K-chunks, A rebuilt from f16 h in LDS (b128 pairs)
    #pragma unroll
    for (int cc = 0; cc < 8; ++cc) {
      const int abase = l16 * 132 + cc * 16 + 4 * half;
      const uint4 a0 = *reinterpret_cast<const uint4*>(&hbuf[abase]);
      const uint4 a1 = *reinterpret_cast<const uint4*>(&hbuf[abase + 8]);
      u32x8 au;
      au[0] = a0.x; au[1] = a0.y; au[2] = a0.z; au[3] = a0.w;
      au[4] = a1.x; au[5] = a1.y; au[6] = a1.z; au[7] = a1.w;
      const v16h afr = __builtin_bit_cast(v16h, au);
      acc[0] = __builtin_amdgcn_wmma_f32_16x16x32_f16(
          false, afr, false, ufr[cc][0], (short)0, acc[0], false, false);
      acc[1] = __builtin_amdgcn_wmma_f32_16x16x32_f16(
          false, afr, false, ufr[cc][1], (short)0, acc[1], false, false);
    }

    // ---- scatter z back into this wave's own columns of xwcur
    //      (wave-private 32-col slice; in-wave DS ordering covers RAW)
    if (half == 0) {
      #pragma unroll
      for (int nt = 0; nt < 2; ++nt) {
        int col = n0 + nt * 16 + l16;
        #pragma unroll
        for (int r = 0; r < 8; ++r)
          xwcur[r * LSTM_4H + col] = acc[nt][r];
      }
    }
    __syncthreads();

    // ---- gates: 2048 elements, 2 per thread (z read from xwcur)
    {
      float c  = cbuf[b0 * LSTM_H + j0];
      float zi = xwcur[b0 * LSTM_4H + j0];
      float zf = xwcur[b0 * LSTM_4H + 256 + j0];
      float zg = xwcur[b0 * LSTM_4H + 512 + j0];
      float zo = xwcur[b0 * LSTM_4H + 768 + j0];
      float ig = sigmoidf_(zi + pi0 * c);
      float fg = sigmoidf_(zf + pf0 * c);
      float g  = tanhf(zg);
      float cn = fg * c + ig * g;
      float og = sigmoidf_(zo + po0 * cn);
      float hn = og * tanhf(cn);
      cbuf[b0 * LSTM_H + j0] = cn;
      hh[b0 * 264 + j0] = (_Float16)hn;
      out[((size_t)b0 * LSTM_T + t) * (2 * LSTM_H) + dir * LSTM_H + j0] = hn;
    }
    {
      float c  = cbuf[b1 * LSTM_H + j1];
      float zi = xwcur[b1 * LSTM_4H + j1];
      float zf = xwcur[b1 * LSTM_4H + 256 + j1];
      float zg = xwcur[b1 * LSTM_4H + 512 + j1];
      float zo = xwcur[b1 * LSTM_4H + 768 + j1];
      float ig = sigmoidf_(zi + pi1 * c);
      float fg = sigmoidf_(zf + pf1 * c);
      float g  = tanhf(zg);
      float cn = fg * c + ig * g;
      float og = sigmoidf_(zo + po1 * cn);
      float hn = og * tanhf(cn);
      cbuf[b1 * LSTM_H + j1] = cn;
      hh[b1 * 264 + j1] = (_Float16)hn;
      out[((size_t)b1 * LSTM_T + t) * (2 * LSTM_H) + dir * LSTM_H + j1] = hn;
    }

    // drain next-step async copy, then make it (and h/c) visible to all waves
    wait_async0();
    __syncthreads();
  }
}

// ---------------------------------------------------------------------------
extern "C" void kernel_launch(void* const* d_in, const int* in_sizes, int n_in,
                              void* d_out, int out_size, void* d_ws, size_t ws_size,
                              hipStream_t stream) {
  const float* x    = (const float*)d_in[0];
  const float* W_f  = (const float*)d_in[1];
  const float* U_f  = (const float*)d_in[2];
  const float* b_f  = (const float*)d_in[3];
  const float* pi_f = (const float*)d_in[4];
  const float* pf_f = (const float*)d_in[5];
  const float* po_f = (const float*)d_in[6];
  const float* W_b  = (const float*)d_in[7];
  const float* U_b  = (const float*)d_in[8];
  const float* b_b  = (const float*)d_in[9];
  const float* pi_b = (const float*)d_in[10];
  const float* pf_b = (const float*)d_in[11];
  const float* po_b = (const float*)d_in[12];
  float* out   = (float*)d_out;
  float* xw_ws = (float*)d_ws;   // needs 2 * 16384 * 1024 * 4 = 128 MB

  dim3 g1(LSTM_4H / 64, LSTM_M / 128, 2);   // 16 x 128 x 2
  xw_gemm<<<g1, 256, 0, stream>>>(x, W_f, b_f, W_b, b_b, xw_ws);

  const size_t dynls = 2u * LSTM_B * LSTM_4H * sizeof(float);  // 64 KB
  lstm_scan<<<dim3(2), dim3(1024), dynls, stream>>>(
      xw_ws, U_f, U_b, pi_f, pf_f, po_f, pi_b, pf_b, po_b, out);
}
